// Graph2Route_pickup_75694503624821
// MI455X (gfx1250) — matrix operand order repaired and checked by reading, not compiled, not run on hardware
//
#include <hip/hip_runtime.h>
#include <hip/hip_bf16.h>

// ---------------------------------------------------------------------------
// Graph2Route pickup forward for MI455X (gfx1250, wave32, WMMA)
// B=8, T=10, N=100, D_H=64, H=72, BT=80, edges rows = 800000
// ---------------------------------------------------------------------------

typedef __attribute__((ext_vector_type(16))) __bf16 v16bf;
typedef __attribute__((ext_vector_type(8)))  float  v8f;

#define NEGV   (-1e9f)

__device__ __forceinline__ __bf16 f2bf(float f) {
  unsigned u = __float_as_uint(f);
  unsigned r = u + 0x7FFFu + ((u >> 16) & 1u);     // round-to-nearest-even
  unsigned short h = (unsigned short)(r >> 16);
  return __builtin_bit_cast(__bf16, h);
}

__device__ __forceinline__ float sigf(float x) { return 1.0f / (1.0f + expf(-x)); }

// ---------------------------------------------------------------------------
// Embedding kernels (tiny K: plain VALU)
// ---------------------------------------------------------------------------
__global__ void embed_nodes_kernel(const float* __restrict__ V,
                                   const float* __restrict__ Wn,  // (64,8)
                                   float* __restrict__ x) {       // (8000,64)
  int idx = blockIdx.x * 256 + threadIdx.x;   // 512000
  int node = idx >> 6, c = idx & 63;
  const float* vr = V + node * 8;
  const float* wr = Wn + c * 8;
  float a = 0.f;
#pragma unroll
  for (int k = 0; k < 8; ++k) a += vr[k] * wr[k];
  x[idx] = a;
}

__global__ void embed_edges_kernel(const float* __restrict__ E,
                                   const float* __restrict__ We,  // (64,5)
                                   float* __restrict__ e) {       // (800000,64)
  long long idx = (long long)blockIdx.x * 256 + threadIdx.x;      // 51,200,000
  int c = (int)(idx & 63);
  long long row = idx >> 6;
  const float* er = E + row * 5;
  const float* wr = We + c * 5;
  float a = 0.f;
#pragma unroll
  for (int k = 0; k < 5; ++k) a += er[k] * wr[k];
  e[idx] = a;
}

// ---------------------------------------------------------------------------
// WMMA GEMM: C(M,64) = A(M,64) @ W(64,64)^T + bias     (bf16 in, f32 acc)
// One wave handles 16 rows; 4 N-tiles x 2 K-chunks of v_wmma_f32_16x16x32_bf16.
// ---------------------------------------------------------------------------
__global__ void __launch_bounds__(256) gemm64_wmma_kernel(
    const float* __restrict__ A, const float* __restrict__ W,
    const float* __restrict__ bias, float* __restrict__ C, int M) {
  int wave = blockIdx.x * 8 + (threadIdx.x >> 5);
  int rowBase = wave * 16;
  if (rowBase >= M) return;
  int lane = threadIdx.x & 31;
  int m = lane & 15, half = lane >> 4;

  const float* arow = A + (size_t)(rowBase + m) * 64;
  __builtin_prefetch(arow + 16 * 64, 0, 1);          // global_prefetch_b8: next tile

  // A layout (16-bit 16x32): elem j -> K = (j<8 ? half*8+j : 16+half*8+(j-8))
  v16bf a0, a1;
#pragma unroll
  for (int j = 0; j < 8; ++j) {
    a0[j]     = f2bf(arow[half * 8 + j]);
    a0[j + 8] = f2bf(arow[16 + half * 8 + j]);
    a1[j]     = f2bf(arow[32 + half * 8 + j]);
    a1[j + 8] = f2bf(arow[48 + half * 8 + j]);
  }
#pragma unroll
  for (int nt = 0; nt < 4; ++nt) {
    int n = nt * 16 + m;
    // B layout (16-bit 32x16): lane -> N = lane&15, elem j -> K = (j<8 ? half*8+j : 16+half*8+(j-8))
    const float* wrow = W + n * 64 + half * 8;       // B[k][n] = W[n][k]
    v16bf b0, b1;
#pragma unroll
    for (int j = 0; j < 16; ++j) { int kk = (j < 8) ? j : j + 8; b0[j] = f2bf(wrow[kk]); b1[j] = f2bf(wrow[32 + kk]); }
    v8f acc = {};
    acc = __builtin_amdgcn_wmma_f32_16x16x32_bf16(false, a0, false, b0, (short)0, acc, false, false);
    acc = __builtin_amdgcn_wmma_f32_16x16x32_bf16(false, a1, false, b1, (short)0, acc, false, false);
    float bv = bias[n];
#pragma unroll
    for (int r = 0; r < 8; ++r) {                    // C: VGPR r -> M = r + 8*half
      int mm = rowBase + r + 8 * half;
      C[(size_t)mm * 64 + n] = acc[r] + bv;
    }
  }
}

// ---------------------------------------------------------------------------
// WMMA GEMM: C(M,72) = A(M,72) @ W(72,72)^T + bias   (K,N padded to 96/80)
// ---------------------------------------------------------------------------
__global__ void __launch_bounds__(256) gemm72_wmma_kernel(
    const float* __restrict__ A, const float* __restrict__ W,
    const float* __restrict__ bias, float* __restrict__ C, int M) {
  int wave = blockIdx.x * 8 + (threadIdx.x >> 5);
  int rowBase = wave * 16;
  if (rowBase >= M) return;
  int lane = threadIdx.x & 31;
  int m = lane & 15, half = lane >> 4;

  const float* arow = A + (size_t)(rowBase + m) * 72;
  v16bf a[3];
#pragma unroll
  for (int kc = 0; kc < 3; ++kc) {
#pragma unroll
    for (int j = 0; j < 16; ++j) {
      int k = kc * 32 + (j < 8 ? half * 8 + j : 16 + half * 8 + (j - 8));
      a[kc][j] = (k < 72) ? f2bf(arow[k]) : f2bf(0.f);
    }
  }
#pragma unroll
  for (int nt = 0; nt < 5; ++nt) {
    int n = nt * 16 + m;
    v8f acc = {};
#pragma unroll
    for (int kc = 0; kc < 3; ++kc) {
      v16bf b;
#pragma unroll
      for (int j = 0; j < 16; ++j) {
        int k = kc * 32 + (j < 8 ? half * 8 + j : 16 + half * 8 + (j - 8));
        b[j] = (n < 72 && k < 72) ? f2bf(W[n * 72 + k]) : f2bf(0.f);
      }
      acc = __builtin_amdgcn_wmma_f32_16x16x32_bf16(false, a[kc], false, b, (short)0, acc, false, false);
    }
    if (n < 72) {
      float bv = bias[n];
#pragma unroll
      for (int r = 0; r < 8; ++r) {
        int mm = rowBase + r + 8 * half;
        C[(size_t)mm * 72 + n] = acc[r] + bv;
      }
    }
  }
}

// ---------------------------------------------------------------------------
// GCN gate + mean aggregation.  One block per (bt,i); etmp updated in place:
// etmp += Vxe[bt,i] + Vxe[bt,j];  agg = sum_j sig(etmp)*Vxn[bt,j] / (sum sig)
// ---------------------------------------------------------------------------
__global__ void __launch_bounds__(256) gate_agg_kernel(
    float* __restrict__ etmp, const float* __restrict__ vxe,
    const float* __restrict__ vxn, const float* __restrict__ ux,
    float* __restrict__ xtmp) {
  __shared__ float s1[256], s2[256];
  int bi = blockIdx.x;                 // bt*100 + i
  int tid = threadIdx.x;
  int c = tid & 63, jg = tid >> 6;
  int bt = bi / 100;
  float vie = vxe[bi * 64 + c];
  float num = 0.f, den = 0.f;
  for (int j = jg; j < 100; j += 4) {
    size_t eidx = ((size_t)bi * 100 + j) * 64 + c;
    float et = etmp[eidx] + vie + vxe[(bt * 100 + j) * 64 + c];
    etmp[eidx] = et;
    float gate = sigf(et);
    num += gate * vxn[(bt * 100 + j) * 64 + c];
    den += gate;
  }
  s1[tid] = num; s2[tid] = den;
  __syncthreads();
  if (jg == 0) {
#pragma unroll
    for (int q = 1; q < 4; ++q) { num += s1[c + 64 * q]; den += s2[c + 64 * q]; }
    xtmp[bi * 64 + c] = ux[bi * 64 + c] + num / (den + 1e-20f);
  }
}

// Deterministic per-channel batchnorm stats (mean, rstd) over `rows` rows x 64ch.
__global__ void __launch_bounds__(256) chan_stats_kernel(
    const float* __restrict__ buf, int rows,
    float* __restrict__ meanOut, float* __restrict__ rstdOut) {
  __shared__ float s1[256], s2[256];
  int c = blockIdx.x, tid = threadIdx.x;
  float s = 0.f, q = 0.f;
  for (int r = tid; r < rows; r += 256) {
    float v = buf[(size_t)r * 64 + c];
    s += v; q += v * v;
  }
  s1[tid] = s; s2[tid] = q;
  __syncthreads();
  for (int st = 128; st > 0; st >>= 1) {
    if (tid < st) { s1[tid] += s1[tid + st]; s2[tid] += s2[tid + st]; }
    __syncthreads();
  }
  if (tid == 0) {
    float me = s1[0] / (float)rows;
    float va = s2[0] / (float)rows - me * me;   // biased variance (torch/jnp)
    meanOut[c] = me;
    rstdOut[c] = rsqrtf(va + 1e-5f);
  }
}

__global__ void edge_update_kernel(float* __restrict__ e, const float* __restrict__ etmp,
                                   const float* __restrict__ mean, const float* __restrict__ rstd,
                                   const float* __restrict__ g, const float* __restrict__ b) {
  long long idx = (long long)blockIdx.x * 256 + threadIdx.x;   // 51,200,000
  int c = (int)(idx & 63);
  float v = (etmp[idx] - mean[c]) * rstd[c] * g[c] + b[c];
  e[idx] += fmaxf(v, 0.f);
}

__global__ void node_update_kernel(float* __restrict__ x, const float* __restrict__ xtmp,
                                   const float* __restrict__ mean, const float* __restrict__ rstd,
                                   const float* __restrict__ g, const float* __restrict__ b) {
  int idx = blockIdx.x * 256 + threadIdx.x;   // 512000
  int c = idx & 63;
  float v = (xtmp[idx] - mean[c]) * rstd[c] * g[c] + b[c];
  x[idx] += fmaxf(v, 0.f);
}

// ---------------------------------------------------------------------------
// GRU over T=10 steps, one block per batch b (8 blocks, independent rows).
// torch gate order r,z,n.  h0 = 0.
// ---------------------------------------------------------------------------
__global__ void __launch_bounds__(256) gru_kernel(
    const float* __restrict__ x,        // (8,10,6400) == GCN node output
    const float* __restrict__ Wih,      // (192,6400)
    const float* __restrict__ Whh,      // (192,64)
    const float* __restrict__ bih, const float* __restrict__ bhh,
    float* __restrict__ gruout) {       // (80,64)
  __shared__ float xs[6400];
  __shared__ float gi[192], gh[192], h[64];
  int b = blockIdx.x, tid = threadIdx.x;
  if (tid < 64) h[tid] = 0.f;
  for (int t = 0; t < 10; ++t) {
    __syncthreads();
    for (int f = tid; f < 6400; f += 256) xs[f] = x[b * 64000 + t * 6400 + f];
    __syncthreads();
    if (tid < 192) {
      float a = bih[tid];
      const float* w1 = Wih + (size_t)tid * 6400;
      for (int f = 0; f < 6400; ++f) a += xs[f] * w1[f];
      gi[tid] = a;
      float g2 = bhh[tid];
      const float* w2 = Whh + tid * 64;
#pragma unroll
      for (int k = 0; k < 64; ++k) g2 += h[k] * w2[k];
      gh[tid] = g2;
    }
    __syncthreads();
    if (tid < 64) {
      float r = sigf(gi[tid] + gh[tid]);
      float z = sigf(gi[64 + tid] + gh[64 + tid]);
      float n = tanhf(gi[128 + tid] + r * gh[128 + tid]);
      float hn = (1.f - z) * n + z * h[tid];
      h[tid] = hn;
      gruout[(b * 10 + t) * 64 + tid] = hn;
    }
  }
}

__global__ void graph_linear_kernel(const float* __restrict__ gruout, // (80,64)
                                    const float* __restrict__ W,      // (6400,64)
                                    const float* __restrict__ bias,   // (6400)
                                    float* __restrict__ nodeh) {      // (80,6400)
  int idx = blockIdx.x * 256 + threadIdx.x;   // 512000
  int bt = idx / 6400, f = idx % 6400;
  const float* g = gruout + bt * 64;
  const float* w = W + (size_t)f * 64;
  float a = bias[f];
#pragma unroll
  for (int k = 0; k < 64; ++k) a += g[k] * w[k];
  nodeh[idx] = a;
}

__global__ void build_emb_kernel(const float* __restrict__ nodeh,  // (80,6400)
                                 const float* __restrict__ V,      // (8000,8)
                                 float* __restrict__ emb) {        // (8000,72)
  int idx = blockIdx.x * 256 + threadIdx.x;   // 576000
  int r = idx / 72, c = idx % 72;
  int bt = r / 100, n = r % 100;
  emb[idx] = (c < 64) ? nodeh[bt * 6400 + n * 64 + c] : V[r * 8 + (c - 64)];
}

// ---------------------------------------------------------------------------
// Pointer-network decoder: one block per BT row (80 blocks), 100 steps, all
// state + ref_g/ref_p LDS-resident.  Writes exp(log_p) and selections.
// ---------------------------------------------------------------------------
__global__ void __launch_bounds__(128) decoder_kernel(
    const float* __restrict__ emb,   // (80,100,72)
    const float* __restrict__ refg,  // (80,100,72)
    const float* __restrict__ refp,  // (80,100,72)
    const float* __restrict__ init_h, const float* __restrict__ init_c,
    const int*   __restrict__ vmask,     // (80,100)
    const int*   __restrict__ start_idx, // (80)
    const float* __restrict__ start_fea, // (80,5)
    const float* __restrict__ snW, const float* __restrict__ snB,  // (72,5),(72)
    const float* __restrict__ cou_fea,    // (8,6)
    const float* __restrict__ worker_emb, // (1000,20)
    const float* __restrict__ lWih, const float* __restrict__ lWhh,   // (288,72)
    const float* __restrict__ lbih, const float* __restrict__ lbhh,   // (288)
    const float* __restrict__ gWq, const float* __restrict__ gbq, const float* __restrict__ gv,
    const float* __restrict__ pWq, const float* __restrict__ pbq, const float* __restrict__ pv,
    float* __restrict__ out) {
  __shared__ float refg_s[7200], refp_s[7200];
  __shared__ float xs[72], hs[72], cs[72], qs[72], q2s[72], qc[97];
  __shared__ float gates[288], us[100], red[128];
  __shared__ int ridx[128], msk[100], curIdx;

  int bt = blockIdx.x, tid = threadIdx.x;
  for (int i2 = tid; i2 < 7200; i2 += 128) {
    refg_s[i2] = refg[bt * 7200 + i2];
    refp_s[i2] = refp[bt * 7200 + i2];
  }
  if (tid < 72) {
    hs[tid] = init_h[bt * 72 + tid];
    cs[tid] = init_c[bt * 72 + tid];
    float a = snB[tid];                       // dec_in0 = start_node_emb(start_fea)
#pragma unroll
    for (int k = 0; k < 5; ++k) a += start_fea[bt * 5 + k] * snW[tid * 5 + k];
    xs[tid] = a;
  }
  int bb = bt / 10;
  if (tid < 20) { int cid = (int)cou_fea[bb * 6]; qc[72 + tid] = worker_emb[cid * 20 + tid]; }
  if (tid >= 20 && tid < 25) qc[72 + tid] = cou_fea[bb * 6 + 1 + (tid - 20)];
  if (tid < 100) msk[tid] = vmask[bt * 100 + tid];
  if (tid == 0) curIdx = start_idx[bt];
  __syncthreads();

  for (int step = 0; step < 100; ++step) {
    // mask visited, then mask_modify (if all masked, unmask last)
    if (tid == 0) msk[curIdx] = 1;
    __syncthreads();
    ridx[tid] = (tid < 100) ? msk[tid] : 1;
    __syncthreads();
    for (int s = 64; s > 0; s >>= 1) { if (tid < s) ridx[tid] &= ridx[tid + s]; __syncthreads(); }
    if (tid == 0 && ridx[0]) msk[99] = 0;
    __syncthreads();

    // LSTM cell (i,f,g,o)
    for (int o = tid; o < 288; o += 128) {
      float a = lbih[o] + lbhh[o];
      const float* w1 = lWih + o * 72;
      const float* w2 = lWhh + o * 72;
      for (int k = 0; k < 72; ++k) a += xs[k] * w1[k] + hs[k] * w2[k];
      gates[o] = a;
    }
    __syncthreads();
    if (tid < 72) {
      float ig = sigf(gates[tid]),       fg = sigf(gates[72 + tid]);
      float gg = tanhf(gates[144 + tid]), og = sigf(gates[216 + tid]);
      float cn = fg * cs[tid] + ig * gg;
      cs[tid] = cn;
      hs[tid] = og * tanhf(cn);
    }
    __syncthreads();

    // glimpse q, scores, softmax, context g
    if (tid < 72) {
      float a = gbq[tid];
      const float* w = gWq + tid * 72;
      for (int k = 0; k < 72; ++k) a += hs[k] * w[k];
      qs[tid] = a;
    }
    __syncthreads();
    if (tid < 100) {
      float a = 0.f;
      const float* rg = refg_s + tid * 72;
      for (int k = 0; k < 72; ++k) a += tanhf(qs[k] + rg[k]) * gv[k];
      us[tid] = msk[tid] ? NEGV : a;
    }
    __syncthreads();
    red[tid] = (tid < 100) ? us[tid] : NEGV;
    __syncthreads();
    for (int s = 64; s > 0; s >>= 1) { if (tid < s) red[tid] = fmaxf(red[tid], red[tid + s]); __syncthreads(); }
    float mx = red[0];
    __syncthreads();
    float ev = (tid < 100) ? expf(us[tid] - mx) : 0.f;
    red[tid] = ev;
    __syncthreads();
    for (int s = 64; s > 0; s >>= 1) { if (tid < s) red[tid] += red[tid + s]; __syncthreads(); }
    float ssum = red[0];
    __syncthreads();
    if (tid < 100) us[tid] = ev / ssum;         // us := softmax probs
    __syncthreads();
    if (tid < 72) {
      float a = 0.f;
      for (int n2 = 0; n2 < 100; ++n2) a += us[n2] * refg_s[n2 * 72 + tid];
      qc[tid] = a;                               // g -> concat buffer
    }
    __syncthreads();

    // pointer q2, scores u2
    if (tid < 72) {
      float a = pbq[tid];
      const float* w = pWq + tid * 97;
      for (int k = 0; k < 97; ++k) a += qc[k] * w[k];
      q2s[tid] = a;
    }
    __syncthreads();
    if (tid < 100) {
      float a = 0.f;
      const float* rp = refp_s + tid * 72;
      for (int k = 0; k < 72; ++k) a += tanhf(q2s[k] + rp[k]) * pv[k];
      us[tid] = msk[tid] ? NEGV : 10.f * tanhf(a);
    }
    __syncthreads();

    // log_softmax -> probs out; argmax -> next index
    red[tid] = (tid < 100) ? us[tid] : NEGV;
    __syncthreads();
    for (int s = 64; s > 0; s >>= 1) { if (tid < s) red[tid] = fmaxf(red[tid], red[tid + s]); __syncthreads(); }
    float mx2 = red[0];
    __syncthreads();
    float e2 = (tid < 100) ? expf(us[tid] - mx2) : 0.f;
    red[tid] = e2;
    __syncthreads();
    for (int s = 64; s > 0; s >>= 1) { if (tid < s) red[tid] += red[tid + s]; __syncthreads(); }
    float s2 = red[0];
    __syncthreads();
    if (tid < 100) out[(bt * 100 + step) * 100 + tid] = e2 / s2;

    red[tid] = (tid < 100) ? us[tid] : (NEGV - 1.f);
    ridx[tid] = tid;
    __syncthreads();
    for (int s = 64; s > 0; s >>= 1) {
      if (tid < s) {
        float a = red[tid], b2 = red[tid + s];
        int ia = ridx[tid], ib = ridx[tid + s];
        if (b2 > a || (b2 == a && ib < ia)) { red[tid] = b2; ridx[tid] = ib; }
      }
      __syncthreads();
    }
    if (tid == 0) { curIdx = ridx[0]; out[800000 + bt * 100 + step] = (float)ridx[0]; }
    __syncthreads();
    if (tid < 72) xs[tid] = emb[(bt * 100 + curIdx) * 72 + tid];
    __syncthreads();
  }
}

// ---------------------------------------------------------------------------
// Host-side orchestration
// ---------------------------------------------------------------------------
extern "C" void kernel_launch(void* const* d_in, const int* in_sizes, int n_in,
                              void* d_out, int out_size, void* d_ws, size_t ws_size,
                              hipStream_t stream) {
  (void)in_sizes; (void)n_in; (void)out_size; (void)ws_size;

  // ---- inputs (setup_inputs dict insertion order, params pytree recursive) --
  const float* V         = (const float*)d_in[0];   // (8,10,100,8)
  const float* E         = (const float*)d_in[1];   // (8,10,100,100,5)
  const int*   Vmask     = (const int*)  d_in[2];   // (8,10,100) bool->int
  const float* start_fea = (const float*)d_in[3];   // (8,10,5)
  const int*   start_idx = (const int*)  d_in[4];   // (8,10)
  const float* cou_fea   = (const float*)d_in[5];   // (8,6)
  const float* init_h    = (const float*)d_in[6];   // (80,72)
  const float* init_c    = (const float*)d_in[7];   // (80,72)
  const float* worker_emb= (const float*)d_in[8];   // (1000,20)
  const float* nodeW     = (const float*)d_in[9];   // (64,8)
  const float* edgeW     = (const float*)d_in[10];  // (64,5)
  const float* snW       = (const float*)d_in[11];  // (72,5)
  const float* snB       = (const float*)d_in[12];  // (72)
  // gcn layer L: base 13 + L*12 :
  //  nU.W,nU.b,nV.W,nV.b,eU.W,eU.b,eV.W,eV.b,bn_ng,bn_nb,bn_eg,bn_eb
  const float* gruWih    = (const float*)d_in[37];  // (192,6400)
  const float* gruWhh    = (const float*)d_in[38];  // (192,64)
  const float* grubih    = (const float*)d_in[39];
  const float* grubhh    = (const float*)d_in[40];
  const float* glW       = (const float*)d_in[41];  // (6400,64)
  const float* glB       = (const float*)d_in[42];  // (6400)
  const float* lWih      = (const float*)d_in[43];  // (288,72)
  const float* lWhh      = (const float*)d_in[44];  // (288,72)
  const float* lbih      = (const float*)d_in[45];
  const float* lbhh      = (const float*)d_in[46];
  const float* gWq       = (const float*)d_in[47];  // (72,72)
  const float* gbq       = (const float*)d_in[48];
  const float* gWref     = (const float*)d_in[49];  // (72,72)
  const float* gbref     = (const float*)d_in[50];
  const float* gv        = (const float*)d_in[51];  // (72)
  const float* pWq       = (const float*)d_in[52];  // (72,97)
  const float* pbq       = (const float*)d_in[53];
  const float* pWref     = (const float*)d_in[54];  // (72,72)
  const float* pbref     = (const float*)d_in[55];
  const float* pv        = (const float*)d_in[56];  // (72)

  float* out = (float*)d_out;

  // ---- workspace layout (floats) ----
  float* w = (float*)d_ws;
  float* e_buf  = w;                    // 51,200,000
  float* etmp   = e_buf  + 51200000;    // 51,200,000
  float* x_buf  = etmp   + 51200000;    // 512,000
  float* xtmp   = x_buf  + 512000;
  float* vxe    = xtmp   + 512000;
  float* vxn    = vxe    + 512000;
  float* uxb    = vxn    + 512000;
  float* emean  = uxb    + 512000;      // 64
  float* erstd  = emean  + 64;
  float* nmean  = erstd  + 64;
  float* nrstd  = nmean  + 64;
  float* gruout = nrstd  + 64;          // 5,120
  float* nodeh  = gruout + 5120;        // 512,000
  float* embb   = nodeh  + 512000;      // 576,000
  float* refg   = embb   + 576000;      // 576,000
  float* refp   = refg   + 576000;      // 576,000

  // ---- embeddings ----
  embed_nodes_kernel<<<2000, 256, 0, stream>>>(V, nodeW, x_buf);
  embed_edges_kernel<<<200000, 256, 0, stream>>>(E, edgeW, e_buf);

  // ---- 2x residual gated GCN layers ----
  for (int L = 0; L < 2; ++L) {
    int base = 13 + L * 12;
    const float* nU_W  = (const float*)d_in[base + 0];
    const float* nU_b  = (const float*)d_in[base + 1];
    const float* nV_W  = (const float*)d_in[base + 2];
    const float* nV_b  = (const float*)d_in[base + 3];
    const float* eU_W  = (const float*)d_in[base + 4];
    const float* eU_b  = (const float*)d_in[base + 5];
    const float* eV_W  = (const float*)d_in[base + 6];
    const float* eV_b  = (const float*)d_in[base + 7];
    const float* bn_ng = (const float*)d_in[base + 8];
    const float* bn_nb = (const float*)d_in[base + 9];
    const float* bn_eg = (const float*)d_in[base + 10];
    const float* bn_eb = (const float*)d_in[base + 11];

    gemm64_wmma_kernel<<<6250, 256, 0, stream>>>(e_buf, eU_W, eU_b, etmp, 800000);
    gemm64_wmma_kernel<<<63, 256, 0, stream>>>(x_buf, eV_W, eV_b, vxe, 8000);
    gemm64_wmma_kernel<<<63, 256, 0, stream>>>(x_buf, nV_W, nV_b, vxn, 8000);
    gemm64_wmma_kernel<<<63, 256, 0, stream>>>(x_buf, nU_W, nU_b, uxb, 8000);
    gate_agg_kernel<<<8000, 256, 0, stream>>>(etmp, vxe, vxn, uxb, xtmp);
    chan_stats_kernel<<<64, 256, 0, stream>>>(etmp, 800000, emean, erstd);
    chan_stats_kernel<<<64, 256, 0, stream>>>(xtmp, 8000, nmean, nrstd);
    edge_update_kernel<<<200000, 256, 0, stream>>>(e_buf, etmp, emean, erstd, bn_eg, bn_eb);
    node_update_kernel<<<2000, 256, 0, stream>>>(x_buf, xtmp, nmean, nrstd, bn_ng, bn_nb);
  }

  // ---- GRU over time + graph linear + decoder context ----
  gru_kernel<<<8, 256, 0, stream>>>(x_buf, gruWih, gruWhh, grubih, grubhh, gruout);
  graph_linear_kernel<<<2000, 256, 0, stream>>>(gruout, glW, glB, nodeh);
  build_emb_kernel<<<2250, 256, 0, stream>>>(nodeh, V, embb);
  gemm72_wmma_kernel<<<63, 256, 0, stream>>>(embb, gWref, gbref, refg, 8000);
  gemm72_wmma_kernel<<<63, 256, 0, stream>>>(embb, pWref, pbref, refp, 8000);

  // ---- pointer decoder (80 independent rows, 100 steps each) ----
  decoder_kernel<<<80, 128, 0, stream>>>(
      embb, refg, refp, init_h, init_c, Vmask, start_idx, start_fea, snW, snB,
      cou_fea, worker_emb, lWih, lWhh, lbih, lbhh,
      gWq, gbq, gv, pWq, pbq, pv, out);
}
